// VQVAE_23407571763616
// MI455X (gfx1250) — matrix operand order repaired and checked by reading, not compiled, not run on hardware
//
#include <hip/hip_runtime.h>
#include <hip/hip_bf16.h>

// ---------- types ----------
typedef __attribute__((ext_vector_type(16))) __bf16         v16bf;
typedef __attribute__((ext_vector_type(8)))  float          v8f;
typedef __attribute__((ext_vector_type(8)))  unsigned short us8;

__device__ __forceinline__ unsigned short f32_to_bf16(float f) {
  unsigned int u = __float_as_uint(f);
  unsigned int r = u + 0x7FFFu + ((u >> 16) & 1u);   // round-to-nearest-even
  return (unsigned short)(r >> 16);
}
__device__ __forceinline__ float bf16_to_f32(unsigned short h) {
  return __uint_as_float(((unsigned int)h) << 16);
}
__device__ __forceinline__ __bf16 us_to_bf(unsigned short u) {
  return __builtin_bit_cast(__bf16, u);
}

// ---------- NCHW f32 -> NHWC bf16 ----------
__global__ void cvt_nchw_to_nhwc_bf16(const float* __restrict__ in,
                                      unsigned short* __restrict__ out,
                                      int C, int H, int W, int total) {
  int i = blockIdx.x * blockDim.x + threadIdx.x;   // index into NHWC output
  if (i >= total) return;
  int p = i / C;           // pixel (b*H*W + h*W + w)
  int c = i - p * C;
  int hw = H * W;
  int b = p / hw;
  int rem = p - b * hw;
  size_t src = (((size_t)b * C + c) * hw) + rem;   // NCHW
  out[i] = f32_to_bf16(in[src]);
}

// ---------- weight packing -> bf16 [Cout][K], K ordered (r, s, ci) with ci fastest ----------
__global__ void prep_weights(const float* __restrict__ w, unsigned short* __restrict__ out,
                             int Cout, int Cin, int kh, int kw, int transposed) {
  int idx = blockIdx.x * blockDim.x + threadIdx.x;
  int K = Cin * kh * kw;
  int total = Cout * K;
  if (idx >= total) return;
  int co  = idx / K;
  int rem = idx - co * K;
  int rs  = rem / Cin;
  int ci  = rem - rs * Cin;
  int r   = rs / kw;
  int s   = rs - r * kw;
  float v;
  if (transposed)  // w layout (Cin, Cout, kh, kw), spatially flipped
    v = w[(((size_t)ci * Cout + co) * kh + (kh - 1 - r)) * kw + (kw - 1 - s)];
  else             // w layout (Cout, Cin, kh, kw)
    v = w[(((size_t)co * Cin + ci) * kh + r) * kw + s];
  out[idx] = f32_to_bf16(v);
}

// ---------- implicit-GEMM conv, NHWC activations, bf16 WMMA, f32 accumulate ----------
// 128-thread blocks = 4 waves; each wave computes a 32(Cout) x 64(pixel) tile
// (2 M-tiles x 4 N-tiles), block covers 256 pixels. Fast path (Cin%32==0):
// division-free incremental K-walk, per-pixel precomputed base addresses,
// two contiguous 16B loads per pixel column per K-chunk.
// Generic path (e1, Cin=4): scalar gather.
// dil=2 + stride=1 implements ConvTranspose2d(stride=2) as an lhs-dilated conv.
// act: 0=none, 1=relu, 2=leaky_relu(0.01).
__global__ __launch_bounds__(128) void conv_wmma_nhwc(
    const unsigned short* __restrict__ X,   // bf16 NHWC
    const unsigned short* __restrict__ Wt,  // bf16 [Cout][K]
    const float* __restrict__ bias,
    const unsigned short* __restrict__ Res, // bf16 NHWC, same shape as out
    unsigned short* __restrict__ Ybf,       // bf16 NHWC out (or null)
    float* __restrict__ Yf32,               // f32 NCHW out (or null)
    int N, int Cin, int Hin, int Win,
    int Cout, int Hout, int Wout,
    int kh, int kw, int stride, int pad, int dil, int act) {
  const int lane = threadIdx.x & 31;       // wave32 lane
  const int wv   = threadIdx.x >> 5;       // wave id within block (0..3)
  const int K    = Cin * kh * kw;
  const int HW   = Hout * Wout;

  const int m  = lane & 15;
  const int hi = lane >> 4;
  const int coT = blockIdx.y * 32;
  const int co_m0 = coT + m;
  const int co_m1 = coT + 16 + m;
  const bool mv0 = co_m0 < Cout;
  const bool mv1 = co_m1 < Cout;
  const unsigned short* wrow0 = Wt + (size_t)co_m0 * (size_t)K;
  const unsigned short* wrow1 = Wt + (size_t)co_m1 * (size_t)K;

  int pix[4], bb[4], oh[4], ow[4], ohs[4], ows[4];
  long long pbase[4];                       // per-pixel NHWC base (elements), dil=1 fast path
#pragma unroll
  for (int t = 0; t < 4; ++t) {
    int p   = blockIdx.x * 256 + wv * 64 + t * 16 + (lane & 15);
    int b   = p / HW;
    int rem = p - b * HW;
    pix[t] = p;
    bb[t]  = b;
    oh[t]  = rem / Wout;
    ow[t]  = rem - oh[t] * Wout;
    ohs[t] = oh[t] * stride - pad;
    ows[t] = ow[t] * stride - pad;
    pbase[t] = (((long long)b * Hin + ohs[t]) * Win + ows[t]) * Cin;
  }

  v8f acc[2][4] = {};

  const int kgA = hi * 16;   // A fragment K groups (contiguous, matches B)
  const int kgB = hi * 16;   // B fragment K groups

  if ((Cin & 31) == 0) {
    // ================= fast path: contiguous channel loads, no divisions =================
    // K order (r, s, ci): lane's 16-block lives in one (r,s) tap since Cin % 32 == 0.
    int rs  = 0;
    int ci0 = kgB;           // kgB in {0,16} < Cin
    int r = 0, s = 0;
    const int rowPitch = Win * Cin;
    for (int k0 = 0; k0 < K; k0 += 32) {
      // A fragments (two Cout tiles)
      us8 w0lo = {}, w0hi = {}, w1lo = {}, w1hi = {};
      if (mv0) {
        w0lo = *(const us8*)(wrow0 + k0 + kgA);
        w0hi = *(const us8*)(wrow0 + k0 + kgA + 8);
        __builtin_prefetch(wrow0 + k0 + 32, 0, 1);
      }
      if (mv1) {
        w1lo = *(const us8*)(wrow1 + k0 + kgA);
        w1hi = *(const us8*)(wrow1 + k0 + kgA + 8);
      }
      v16bf a0, a1;
#pragma unroll
      for (int i = 0; i < 8; ++i) {
        a0[i] = us_to_bf(w0lo[i]); a0[8 + i] = us_to_bf(w0hi[i]);
        a1[i] = us_to_bf(w1lo[i]); a1[8 + i] = us_to_bf(w1hi[i]);
      }

      const int tapOff = r * rowPitch + s * Cin + ci0;   // element offset of this tap

      v16bf bv[4];
#pragma unroll
      for (int t = 0; t < 4; ++t) {
        int ihd = ohs[t] + r;
        int iwd = ows[t] + s;
        us8 lo = {}, hi8 = {};
        if (dil == 1) {
          bool ok = (ihd >= 0) & (iwd >= 0) & (ihd < Hin) & (iwd < Win);
          if (ok) {
            const unsigned short* px = X + (pbase[t] + tapOff);
            lo  = *(const us8*)px;
            hi8 = *(const us8*)(px + 8);
          }
        } else {  // dil == 2 (stride == 1)
          bool ok = (ihd >= 0) & (iwd >= 0) & !(ihd & 1) & !(iwd & 1);
          int ih = ihd >> 1, iw = iwd >> 1;
          ok = ok & (ih < Hin) & (iw < Win);
          if (ok) {
            const unsigned short* px =
                X + (((size_t)bb[t] * Hin + ih) * Win + iw) * Cin + ci0;
            lo  = *(const us8*)px;
            hi8 = *(const us8*)(px + 8);
          }
        }
#pragma unroll
        for (int i = 0; i < 8; ++i) { bv[t][i] = us_to_bf(lo[i]); bv[t][8 + i] = us_to_bf(hi8[i]); }
      }

#pragma unroll
      for (int t = 0; t < 4; ++t) {
        acc[0][t] = __builtin_amdgcn_wmma_f32_16x16x32_bf16(
            false, a0, false, bv[t], (short)0, acc[0][t], false, false);
        acc[1][t] = __builtin_amdgcn_wmma_f32_16x16x32_bf16(
            false, a1, false, bv[t], (short)0, acc[1][t], false, false);
      }

      // incremental K walk (no divisions)
      ci0 += 32;
      if (ci0 >= Cin) {
        ci0 -= Cin;
        ++rs;
        if (++s == kw) { s = 0; ++r; }
      }
    }
  } else {
    // ================= generic path (Cin not multiple of 32; only e1) =================
    for (int k0 = 0; k0 < K; k0 += 32) {
      us8 w0lo = {}, w0hi = {}, w1lo = {}, w1hi = {};
      if (mv0) {
        w0lo = *(const us8*)(wrow0 + k0 + kgA);
        w0hi = *(const us8*)(wrow0 + k0 + kgA + 8);
      }
      if (mv1) {
        w1lo = *(const us8*)(wrow1 + k0 + kgA);
        w1hi = *(const us8*)(wrow1 + k0 + kgA + 8);
      }
      v16bf a0, a1;
#pragma unroll
      for (int i = 0; i < 8; ++i) {
        a0[i] = us_to_bf(w0lo[i]); a0[8 + i] = us_to_bf(w0hi[i]);
        a1[i] = us_to_bf(w1lo[i]); a1[8 + i] = us_to_bf(w1hi[i]);
      }

      // K order: (r, s, ci) with ci fastest
      int kk = k0 + kgB;
      int rs = kk / Cin;
      int ci = kk - rs * Cin;
      int r  = rs / kw;
      int s  = rs - r * kw;

      v16bf bv[4];
#pragma unroll
      for (int i = 0; i < 16; ++i) {
#pragma unroll
        for (int t = 0; t < 4; ++t) {
          int ihd = ohs[t] + r;
          int iwd = ows[t] + s;
          unsigned short val = 0;
          if (dil == 1) {
            if (ihd >= 0 && iwd >= 0 && ihd < Hin && iwd < Win)
              val = X[(((size_t)bb[t] * Hin + ihd) * Win + iwd) * Cin + ci];
          } else {
            if (ihd >= 0 && iwd >= 0 && !(ihd & 1) && !(iwd & 1)) {
              int ih = ihd >> 1, iw = iwd >> 1;
              if (ih < Hin && iw < Win)
                val = X[(((size_t)bb[t] * Hin + ih) * Win + iw) * Cin + ci];
            }
          }
          bv[t][i] = us_to_bf(val);
        }
        if (++ci == Cin) { ci = 0; if (++s == kw) { s = 0; if (++r == kh) r = 0; } }
      }

#pragma unroll
      for (int t = 0; t < 4; ++t) {
        acc[0][t] = __builtin_amdgcn_wmma_f32_16x16x32_bf16(
            false, a0, false, bv[t], (short)0, acc[0][t], false, false);
        acc[1][t] = __builtin_amdgcn_wmma_f32_16x16x32_bf16(
            false, a1, false, bv[t], (short)0, acc[1][t], false, false);
      }
    }
  }

  // ---- epilogue: VGPR j holds M=j (lanes 0-15) / M=j+8 (lanes 16-31), N=lane&15 ----
  const int mrow = hi << 3;
#pragma unroll
  for (int mt = 0; mt < 2; ++mt) {
#pragma unroll
    for (int t = 0; t < 4; ++t) {
#pragma unroll
      for (int j = 0; j < 8; ++j) {
        int co = coT + mt * 16 + mrow + j;
        if (co < Cout) {
          float v = acc[mt][t][j];
          if (bias) v += bias[co];
          size_t oidx = (size_t)pix[t] * Cout + co;     // NHWC
          if (Res) v += bf16_to_f32(Res[oidx]);
          if (act == 1)      v = fmaxf(v, 0.0f);
          else if (act == 2) v = (v > 0.0f) ? v : 0.01f * v;
          if (Ybf) Ybf[oidx] = f32_to_bf16(v);
          if (Yf32) {                                    // NCHW (reference output layout)
            size_t onchw = (((size_t)bb[t] * Cout + co) * Hout + oh[t]) * Wout + ow[t];
            Yf32[onchw] = v;
          }
        }
      }
    }
  }
}

// ---------- zero the loss scalar ----------
__global__ void zero_scalar(float* p) {
  if (threadIdx.x == 0 && blockIdx.x == 0) *p = 0.0f;
}

// ---------- VQ bottleneck (NHWC z/q: channels contiguous) ----------
__global__ __launch_bounds__(256) void vq_kernel(
    const unsigned short* __restrict__ z,   // bf16 [NP][64]
    const float* __restrict__ cb,           // f32 [512][64]
    unsigned short* __restrict__ q,         // bf16 [NP][64]
    float* __restrict__ idx_out,            // f32 [NP]
    float* __restrict__ loss_out,           // scalar accumulator
    int NP, float scale) {
  constexpr int D = 64, KC = 512;
  int p = blockIdx.x * blockDim.x + threadIdx.x;
  float lsum = 0.0f;
  if (p < NP) {
    size_t base = (size_t)p * D;
    float zv[D];
#pragma unroll
    for (int d = 0; d < D; ++d) zv[d] = bf16_to_f32(z[base + d]);
    float best = 3.4e38f;
    int bi = 0;
    for (int k = 0; k < KC; ++k) {
      const float* c = cb + (size_t)k * D;
      float dist = 0.0f;
#pragma unroll
      for (int d = 0; d < D; ++d) { float t = zv[d] - c[d]; dist = fmaf(t, t, dist); }
      if (dist < best) { best = dist; bi = k; }
    }
    idx_out[p] = (float)bi;
    const float* c = cb + (size_t)bi * D;
#pragma unroll
    for (int d = 0; d < D; ++d) {
      float cq = c[d];
      q[base + d] = f32_to_bf16(cq);        // straight-through forward value is q
      float df = cq - zv[d];
      lsum = fmaf(df, df, lsum);
    }
  }
  __shared__ float red[256];
  red[threadIdx.x] = lsum;
  __syncthreads();
  for (int off = 128; off > 0; off >>= 1) {
    if ((int)threadIdx.x < off) red[threadIdx.x] += red[threadIdx.x + off];
    __syncthreads();
  }
  if (threadIdx.x == 0) atomicAdd(loss_out, red[0] * scale);
}

// ---------- host side ----------
extern "C" void kernel_launch(void* const* d_in, const int* in_sizes, int n_in,
                              void* d_out, int out_size, void* d_ws, size_t ws_size,
                              hipStream_t stream) {
  (void)in_sizes; (void)n_in; (void)out_size; (void)ws_size;
  const float* x_in     = (const float*)d_in[0];
  const float* e1_w     = (const float*)d_in[1];  const float* e1_b   = (const float*)d_in[2];
  const float* e2_w     = (const float*)d_in[3];  const float* e2_b   = (const float*)d_in[4];
  const float* e3_w     = (const float*)d_in[5];  const float* e3_b   = (const float*)d_in[6];
  const float* er1_w1   = (const float*)d_in[7];  const float* er1_w2 = (const float*)d_in[8];
  const float* er2_w1   = (const float*)d_in[9];  const float* er2_w2 = (const float*)d_in[10];
  const float* e4_w     = (const float*)d_in[11]; const float* e4_b   = (const float*)d_in[12];
  const float* codebook = (const float*)d_in[13];
  const float* d1_w     = (const float*)d_in[14]; const float* d1_b   = (const float*)d_in[15];
  const float* dr1_w1   = (const float*)d_in[16]; const float* dr1_w2 = (const float*)d_in[17];
  const float* dr2_w1   = (const float*)d_in[18]; const float* dr2_w2 = (const float*)d_in[19];
  const float* dt1_w    = (const float*)d_in[20]; const float* dt1_b  = (const float*)d_in[21];
  const float* dt2_w    = (const float*)d_in[22]; const float* dt2_b  = (const float*)d_in[23];

  float* y_out   = (float*)d_out;                  // [16,4,256,256] NCHW
  float* idx_out = y_out + 16 * 4 * 256 * 256;     // [16,64,64]
  float* loss    = idx_out + 16 * 64 * 64;         // scalar

  // ---- workspace carve-out (256B aligned) ----
  char* ws = (char*)d_ws;
  size_t off = 0;
  auto alloc = [&](size_t bytes) -> char* {
    char* p = ws + off;
    off += (bytes + 255) & ~(size_t)255;
    return p;
  };
  auto wbuf = [&](int elems) -> unsigned short* {
    return (unsigned short*)alloc((size_t)elems * 2);
  };

  unsigned short* wE1  = wbuf(64 * 4 * 16);
  unsigned short* wE2  = wbuf(128 * 64 * 16);
  unsigned short* wE3  = wbuf(128 * 128 * 9);
  unsigned short* wR1a = wbuf(128 * 128 * 9);
  unsigned short* wR1b = wbuf(128 * 128);
  unsigned short* wR2a = wbuf(128 * 128 * 9);
  unsigned short* wR2b = wbuf(128 * 128);
  unsigned short* wE4  = wbuf(64 * 128);
  unsigned short* wD1  = wbuf(128 * 64 * 9);
  unsigned short* wS1a = wbuf(128 * 128 * 9);
  unsigned short* wS1b = wbuf(128 * 128);
  unsigned short* wS2a = wbuf(128 * 128 * 9);
  unsigned short* wS2b = wbuf(128 * 128);
  unsigned short* wT1  = wbuf(64 * 128 * 16);
  unsigned short* wT2  = wbuf(4 * 64 * 16);

  unsigned short* B0 = wbuf(16 * 64 * 128 * 128);   // 33.5 MB ping
  unsigned short* B1 = wbuf(16 * 64 * 128 * 128);   // 33.5 MB pong
  unsigned short* B2 = wbuf(16 * 128 * 64 * 64);    // 16.8 MB res branch

  // ---- pack weights to bf16 [Cout][(r,s,ci)] ----
  auto prep = [&](const float* w, unsigned short* o, int Cout, int Cin, int kh, int kw, int tr) {
    int total = Cout * Cin * kh * kw;
    prep_weights<<<(total + 255) / 256, 256, 0, stream>>>(w, o, Cout, Cin, kh, kw, tr);
  };
  prep(e1_w,  wE1,  64, 4, 4, 4, 0);
  prep(e2_w,  wE2, 128, 64, 4, 4, 0);
  prep(e3_w,  wE3, 128, 128, 3, 3, 0);
  prep(er1_w1,wR1a,128, 128, 3, 3, 0);
  prep(er1_w2,wR1b,128, 128, 1, 1, 0);
  prep(er2_w1,wR2a,128, 128, 3, 3, 0);
  prep(er2_w2,wR2b,128, 128, 1, 1, 0);
  prep(e4_w,  wE4,  64, 128, 1, 1, 0);
  prep(d1_w,  wD1, 128, 64, 3, 3, 0);
  prep(dr1_w1,wS1a,128, 128, 3, 3, 0);
  prep(dr1_w2,wS1b,128, 128, 1, 1, 0);
  prep(dr2_w1,wS2a,128, 128, 3, 3, 0);
  prep(dr2_w2,wS2b,128, 128, 1, 1, 0);
  prep(dt1_w, wT1,  64, 128, 4, 4, 1);   // ConvTranspose: flip + IO->OI transpose
  prep(dt2_w, wT2,   4, 64, 4, 4, 1);

  auto conv = [&](const unsigned short* X, const unsigned short* Wt, const float* bias,
                  const unsigned short* Res, unsigned short* Ybf, float* Yf32,
                  int N, int Cin, int Hin, int Win, int Cout, int Hout, int Wout,
                  int kh, int kw, int stride, int pad, int dil, int act) {
    int NP = N * Hout * Wout;                 // always a multiple of 256 in this net
    dim3 grid(NP / 256, (Cout + 31) / 32);
    conv_wmma_nhwc<<<grid, 128, 0, stream>>>(X, Wt, bias, Res, Ybf, Yf32,
                                             N, Cin, Hin, Win, Cout, Hout, Wout,
                                             kh, kw, stride, pad, dil, act);
  };

  // ---- encoder (all activations NHWC bf16) ----
  cvt_nchw_to_nhwc_bf16<<<(16 * 4 * 256 * 256 + 255) / 256, 256, 0, stream>>>(
      x_in, B0, 4, 256, 256, 16 * 4 * 256 * 256);
  conv(B0, wE1, e1_b, nullptr, B1, nullptr, 16, 4, 256, 256, 64, 128, 128, 4, 4, 2, 1, 1, 2);
  conv(B1, wE2, e2_b, nullptr, B0, nullptr, 16, 64, 128, 128, 128, 64, 64, 4, 4, 2, 1, 1, 2);
  conv(B0, wE3, e3_b, nullptr, B1, nullptr, 16, 128, 64, 64, 128, 64, 64, 3, 3, 1, 1, 1, 2);
  // res block 1: h = relu(conv3x3(x)); x = x + conv1x1(h)
  conv(B1, wR1a, nullptr, nullptr, B2, nullptr, 16, 128, 64, 64, 128, 64, 64, 3, 3, 1, 1, 1, 1);
  conv(B2, wR1b, nullptr, B1, B0, nullptr, 16, 128, 64, 64, 128, 64, 64, 1, 1, 1, 0, 1, 0);
  // res block 2 (+ fused trailing lrelu)
  conv(B0, wR2a, nullptr, nullptr, B2, nullptr, 16, 128, 64, 64, 128, 64, 64, 3, 3, 1, 1, 1, 1);
  conv(B2, wR2b, nullptr, B0, B1, nullptr, 16, 128, 64, 64, 128, 64, 64, 1, 1, 1, 0, 1, 2);
  // e4 1x1 -> z (lrelu)
  conv(B1, wE4, e4_b, nullptr, B0, nullptr, 16, 128, 64, 64, 64, 64, 64, 1, 1, 1, 0, 1, 2);

  // ---- VQ bottleneck ----
  zero_scalar<<<1, 1, 0, stream>>>(loss);
  {
    int NP = 16 * 64 * 64;
    float scale = 0.25f / ((float)NP * 64.0f);     // beta * mean over B*H*W*D
    vq_kernel<<<NP / 256, 256, 0, stream>>>(B0, codebook, B1, idx_out, loss, NP, scale);
  }

  // ---- decoder ----
  conv(B1, wD1, d1_b, nullptr, B0, nullptr, 16, 64, 64, 64, 128, 64, 64, 3, 3, 1, 1, 1, 2);
  conv(B0, wS1a, nullptr, nullptr, B2, nullptr, 16, 128, 64, 64, 128, 64, 64, 3, 3, 1, 1, 1, 1);
  conv(B2, wS1b, nullptr, B0, B1, nullptr, 16, 128, 64, 64, 128, 64, 64, 1, 1, 1, 0, 1, 0);
  conv(B1, wS2a, nullptr, nullptr, B2, nullptr, 16, 128, 64, 64, 128, 64, 64, 3, 3, 1, 1, 1, 1);
  conv(B2, wS2b, nullptr, B1, B0, nullptr, 16, 128, 64, 64, 128, 64, 64, 1, 1, 1, 0, 1, 2);
  // ConvTranspose2d as lhs-dilated conv: dil=2, pad=k-1-1=2, stride=1
  conv(B0, wT1, dt1_b, nullptr, B1, nullptr, 16, 128, 64, 64, 64, 128, 128, 4, 4, 1, 2, 2, 2);
  conv(B1, wT2, dt2_b, nullptr, nullptr, y_out, 16, 64, 128, 128, 4, 256, 256, 4, 4, 1, 2, 2, 1);
}
